// TemeralRGAT_55748675502979
// MI455X (gfx1250) — matrix operand order repaired and checked by reading, not compiled, not run on hardware
//
#include <hip/hip_runtime.h>

// ---------------------------------------------------------------------------
// Types / helpers
// ---------------------------------------------------------------------------
typedef unsigned short u16t;
typedef __attribute__((ext_vector_type(16))) __bf16 v16bf;
typedef __attribute__((ext_vector_type(8)))  float   v8f;
typedef __attribute__((ext_vector_type(4)))  int     v4i;

struct __align__(16) U4 { unsigned x, y, z, w; };
union BFrag { v16bf v; U4 q[2]; };

__device__ __forceinline__ u16t f2bf(float f) {
  unsigned u = __float_as_uint(f);
  unsigned r = u + 0x7FFFu + ((u >> 16) & 1u);   // round-to-nearest-even
  return (u16t)(r >> 16);
}
__device__ __forceinline__ float lrelu(float x) { return x > 0.f ? x : 0.2f * x; }

// atomic float max via signed/unsigned int ordering trick (init to -inf)
__device__ __forceinline__ void atomicMaxF(float* addr, float v) {
  if (v >= 0.f) atomicMax((int*)addr, __float_as_int(v));
  else          atomicMin((unsigned int*)addr, __float_as_uint(v));
}

// ---- gfx1250 async copy to LDS (ASYNCcnt path), feature-guarded ----
#if defined(__has_builtin)
#  if __has_builtin(__builtin_amdgcn_global_load_async_to_lds_b128)
#    define USE_ASYNC 1
#  endif
#endif
#ifndef USE_ASYNC
#  define USE_ASYNC 0
#endif

#if USE_ASYNC
typedef __attribute__((address_space(1))) v4i* gv4p;   // global (AS1) int4*
typedef __attribute__((address_space(3))) v4i* lv4p;   // LDS (AS3) int4*

__device__ __forceinline__ void asyncCopy16(const u16t* g, u16t* l) {
  __builtin_amdgcn_global_load_async_to_lds_b128((gv4p)g, (lv4p)l, 0, 0);
}
template <int N>
__device__ __forceinline__ void waitAsync() {
#if __has_builtin(__builtin_amdgcn_s_wait_asynccnt)
  __builtin_amdgcn_s_wait_asynccnt(N);
#else
  asm volatile("s_wait_asynccnt %0" :: "i"(N) : "memory");
#endif
}
#endif

// ---------------------------------------------------------------------------
// WMMA bf16 GEMM: C[M,N] = A[M,K](bf16,row-major) @ Bt[N,K](bf16, PRE-TRANSPOSED)
// Block tile 128x128, 8 wave32 waves (4x2), 32x64 per wave, K-step 32.
// Double-buffered LDS; async-to-LDS pipeline when available.
// ---------------------------------------------------------------------------
#define BM 128
#define BN 128
#define KS 32
#define KPAD 40

__global__ __launch_bounds__(256)
void k_gemm_bf16(const u16t* __restrict__ A, const u16t* __restrict__ Bt,
                 const float* __restrict__ bias, float* __restrict__ C,
                 int M, int N, int K)
{
  __shared__ __align__(16) u16t As[2][BM][KPAD];   // [m][k]
  __shared__ __align__(16) u16t Bs[2][BN][KPAD];   // [n][k]

  const int tid   = threadIdx.x;
  const int lane  = tid & 31;
  const int half  = lane >> 4;
  const int l16   = lane & 15;
  const int wave  = tid >> 5;
  const int waveM = (wave & 3) * 32;
  const int waveN = (wave >> 2) * 64;

  const int blockM = blockIdx.y * BM;
  const int blockN = blockIdx.x * BN;

  // staging coords: each thread moves 16 bf16 (2 x b128) of A and of Bt
  const int srow = tid >> 1;          // 0..127
  const int scol = (tid & 1) * 16;    // 0 or 16
  int gr = blockM + srow; if (gr >= M) gr = M - 1;   // clamp M tail (rows >= M never stored)
  const int gbn = blockN + srow;                     // N % 128 == 0, always valid

  v8f acc[2][4] = {};

#if !USE_ASYNC
  U4 ra0, ra1, rb0, rb1;              // register-pipelined staging
#endif

  auto stageIssue = [&](int buf, int kk) {
    const u16t* ga = A  + (size_t)gr  * K + kk + scol;
    const u16t* gb = Bt + (size_t)gbn * K + kk + scol;
#if USE_ASYNC
    asyncCopy16(ga,     &As[buf][srow][scol]);
    asyncCopy16(ga + 8, &As[buf][srow][scol + 8]);
    asyncCopy16(gb,     &Bs[buf][srow][scol]);
    asyncCopy16(gb + 8, &Bs[buf][srow][scol + 8]);
#else
    (void)buf;
    const U4* pa = (const U4*)ga;
    const U4* pb = (const U4*)gb;
    ra0 = pa[0]; ra1 = pa[1];
    rb0 = pb[0]; rb1 = pb[1];
#endif
  };
#if !USE_ASYNC
  auto stageCommit = [&](int buf) {
    *(U4*)&As[buf][srow][scol]     = ra0;
    *(U4*)&As[buf][srow][scol + 8] = ra1;
    *(U4*)&Bs[buf][srow][scol]     = rb0;
    *(U4*)&Bs[buf][srow][scol + 8] = rb1;
  };
#endif

  auto computeTile = [&](int cur) {
    BFrag af[2];
#pragma unroll
    for (int i = 0; i < 2; ++i) {
      int r = waveM + i * 16 + l16;
      af[i].q[0] = *(const U4*)&As[cur][r][half * 8];        // K = 8*half + 0..7
      af[i].q[1] = *(const U4*)&As[cur][r][16 + half * 8];   // K = 16+8*half + 0..7
    }
#pragma unroll
    for (int j = 0; j < 4; ++j) {
      BFrag bf;
      int n = waveN + j * 16 + l16;
      bf.q[0] = *(const U4*)&Bs[cur][n][half * 8];           // K = 8*half + 0..7
      bf.q[1] = *(const U4*)&Bs[cur][n][16 + half * 8];      // K = 16+8*half + 0..7
#pragma unroll
      for (int i = 0; i < 2; ++i)
        acc[i][j] = __builtin_amdgcn_wmma_f32_16x16x32_bf16(
            false, af[i].v, false, bf.v, (short)0, acc[i][j], false, false);
    }
  };

  const int nk = K / KS;
#if USE_ASYNC
  stageIssue(0, 0);
  for (int ik = 0; ik < nk; ++ik) {
    const int cur = ik & 1;
    const bool nxt = (ik + 1) < nk;
    if (nxt) stageIssue(1 - cur, (ik + 1) * KS);   // prev readers of buf(1-cur) passed last barrier
    if (nxt) waitAsync<4>(); else waitAsync<0>();  // my 4 copies for `cur` are complete (in-order)
    __syncthreads();                               // everyone's copies visible
    computeTile(cur);
    __syncthreads();
  }
#else
  stageIssue(0, 0);
  stageCommit(0);
  __syncthreads();
  for (int ik = 0; ik < nk; ++ik) {
    const int cur = ik & 1;
    const bool nxt = (ik + 1) < nk;
    if (nxt) stageIssue(0, (ik + 1) * KS);         // global loads in flight during compute
    computeTile(cur);
    __syncthreads();
    if (nxt) { stageCommit(1 - cur); __syncthreads(); }
  }
#endif

  // ---- writeback (C layout: vgpr r -> row r + 8*half, col = lane%16) ----
#pragma unroll
  for (int i = 0; i < 2; ++i) {
#pragma unroll
    for (int j = 0; j < 4; ++j) {
      int gcol = blockN + waveN + j * 16 + l16;
      float bv = bias ? bias[gcol] : 0.f;
#pragma unroll
      for (int r = 0; r < 8; ++r) {
        int grow = blockM + waveM + i * 16 + half * 8 + r;
        if (grow < M) C[(size_t)grow * N + gcol] = acc[i][j][r] + bv;
      }
    }
  }
}

// ---------------------------------------------------------------------------
// Elementwise utilities
// ---------------------------------------------------------------------------
__global__ void k_f2bf(const float* __restrict__ s, u16t* __restrict__ d, long n) {
  long i = (long)blockIdx.x * blockDim.x + threadIdx.x;
  long st = (long)gridDim.x * blockDim.x;
  for (; i < n; i += st) d[i] = f2bf(s[i]);
}

// convert + transpose: src f32 [K,N] row-major -> dst bf16 [N,K] row-major
__global__ void k_f2bf_t(const float* __restrict__ s, u16t* __restrict__ d, int K, int N) {
  long tot = (long)K * N;
  long i = (long)blockIdx.x * blockDim.x + threadIdx.x;
  long st = (long)gridDim.x * blockDim.x;
  for (; i < tot; i += st) {
    int k = (int)(i / N);
    int n = (int)(i - (long)k * N);
    d[(size_t)n * K + k] = f2bf(s[i]);
  }
}

__global__ void k_fill_u32(unsigned* __restrict__ p, unsigned v, long n) {
  long i = (long)blockIdx.x * blockDim.x + threadIdx.x;
  long st = (long)gridDim.x * blockDim.x;
  for (; i < n; i += st) p[i] = v;
}

__global__ void k_gap(const int* __restrict__ row, const int* __restrict__ col,
                      const int* __restrict__ years, float* __restrict__ gap, int E) {
  int e = blockIdx.x * blockDim.x + threadIdx.x;
  if (e < E) {
    float d = fabsf((float)years[row[e]] - (float)years[col[e]]);
    gap[e] = expf(-d);
  }
}

// ---------------------------------------------------------------------------
// Temporal attention: per-edge, per-channel segment softmax in 3 passes
// ---------------------------------------------------------------------------
__global__ __launch_bounds__(256)
void k_temporal_max(const float* __restrict__ x1, const float* __restrict__ x2,
                    const float* __restrict__ att1, const float* __restrict__ att2,
                    const int* __restrict__ row, const int* __restrict__ col,
                    const float* __restrict__ gap, float* __restrict__ m, int cin) {
  int e = blockIdx.x;
  int r = row[e], c = col[e];
  float g = gap[e];
  const float* p1 = x1 + (size_t)c * cin;
  const float* p2 = x2 + (size_t)r * cin;
  float* pm = m + (size_t)r * cin;
  for (int i = threadIdx.x; i < cin; i += 256) {
    float a = lrelu(g * (p1[i] * att1[i] + p2[i] * att2[i]));
    atomicMaxF(&pm[i], a);
  }
}

__global__ __launch_bounds__(256)
void k_temporal_sum(const float* __restrict__ x1, const float* __restrict__ x2,
                    const float* __restrict__ att1, const float* __restrict__ att2,
                    const int* __restrict__ row, const int* __restrict__ col,
                    const float* __restrict__ gap, const float* __restrict__ m,
                    float* __restrict__ s, int cin) {
  int e = blockIdx.x;
  int r = row[e], c = col[e];
  float g = gap[e];
  const float* p1 = x1 + (size_t)c * cin;
  const float* p2 = x2 + (size_t)r * cin;
  const float* pm = m + (size_t)r * cin;
  float* ps = s + (size_t)r * cin;
  for (int i = threadIdx.x; i < cin; i += 256) {
    float a = lrelu(g * (p1[i] * att1[i] + p2[i] * att2[i]));
    atomicAdd(&ps[i], expf(a - pm[i]));
  }
}

__global__ __launch_bounds__(256)
void k_temporal_agg(const float* __restrict__ x1, const float* __restrict__ x2,
                    const float* __restrict__ att1, const float* __restrict__ att2,
                    const int* __restrict__ row, const int* __restrict__ col,
                    const float* __restrict__ gap, const float* __restrict__ m,
                    const float* __restrict__ s, float* __restrict__ outT, int cin) {
  __shared__ float red[256];
  int e = blockIdx.x;
  int r = row[e], c = col[e];
  float g = gap[e];
  const float* p1 = x1 + (size_t)c * cin;
  const float* p2 = x2 + (size_t)r * cin;
  const float* pm = m + (size_t)r * cin;
  const float* ps = s + (size_t)r * cin;
  float xv[4];
  float p = 0.f;
  const int iters = cin >> 8;           // cin in {768,1024}
  for (int it = 0; it < iters; ++it) {
    int i = it * 256 + threadIdx.x;
    float x1v = p1[i];
    xv[it] = x1v;
    float a = lrelu(g * (x1v * att1[i] + p2[i] * att2[i]));
    p += expf(a - pm[i]) / ps[i];
  }
  red[threadIdx.x] = p; __syncthreads();
  for (int st = 128; st > 0; st >>= 1) {
    if (threadIdx.x < st) red[threadIdx.x] += red[threadIdx.x + st];
    __syncthreads();
  }
  float w = red[0];
  float* po = outT + (size_t)r * cin;
  for (int it = 0; it < iters; ++it) {
    int i = it * 256 + threadIdx.x;
    atomicAdd(&po[i], xv[it] * w);
  }
}

// ---------------------------------------------------------------------------
// GAT: head-dot reduction, per-(edge,head) segment softmax, scatter-aggregate
// ---------------------------------------------------------------------------
__global__ __launch_bounds__(256)
void k_attn_reduce(const float* __restrict__ hbuf, const float* __restrict__ att,
                   float* __restrict__ aout) {
  __shared__ float red[256];
  int v = blockIdx.x;
  const float* hv = hbuf + (size_t)v * 1024;
#pragma unroll
  for (int h = 0; h < 4; ++h) {
    float p = hv[h * 256 + threadIdx.x] * att[h * 256 + threadIdx.x];
    __syncthreads();
    red[threadIdx.x] = p; __syncthreads();
    for (int st = 128; st > 0; st >>= 1) {
      if (threadIdx.x < st) red[threadIdx.x] += red[threadIdx.x + st];
      __syncthreads();
    }
    if (threadIdx.x == 0) aout[v * 4 + h] = red[0];
  }
}

__global__ void k_gat_logits(const float* __restrict__ asrc, const float* __restrict__ adst,
                             const int* __restrict__ row, const int* __restrict__ col,
                             float* __restrict__ ebuf, float* __restrict__ m2, int E) {
  int idx = blockIdx.x * blockDim.x + threadIdx.x;
  if (idx >= E * 4) return;
  int e = idx >> 2, h = idx & 3;
  float v = lrelu(asrc[col[e] * 4 + h] + adst[row[e] * 4 + h]);
  ebuf[idx] = v;
  atomicMaxF(&m2[row[e] * 4 + h], v);
}

__global__ void k_gat_exp(float* __restrict__ ebuf, const int* __restrict__ row,
                          const float* __restrict__ m2, float* __restrict__ s2, int E) {
  int idx = blockIdx.x * blockDim.x + threadIdx.x;
  if (idx >= E * 4) return;
  int e = idx >> 2, h = idx & 3;
  float ex = expf(ebuf[idx] - m2[row[e] * 4 + h]);
  ebuf[idx] = ex;
  atomicAdd(&s2[row[e] * 4 + h], ex);
}

__global__ __launch_bounds__(256)
void k_gat_scatter(const float* __restrict__ hsrc, const float* __restrict__ ebuf,
                   const float* __restrict__ s2, const int* __restrict__ row,
                   const int* __restrict__ col, float* __restrict__ out, int E) {
  int e = blockIdx.x;
  int r = row[e], c = col[e];
  const float* hs = hsrc + (size_t)c * 1024;
  float* po = out + (size_t)r * 1024;
#pragma unroll
  for (int it = 0; it < 4; ++it) {
    float alpha = ebuf[e * 4 + it] / s2[r * 4 + it];
    int i = it * 256 + threadIdx.x;
    atomicAdd(&po[i], hs[i] * alpha);
  }
}

// ---------------------------------------------------------------------------
// Fused: + sum_j gat_b[j], * gamma/sqrt(1+eps), + beta, ELU; emit f32 and/or bf16
// ---------------------------------------------------------------------------
__global__ void k_bn_elu(const float* __restrict__ acc, const float* __restrict__ gb,
                         const float* __restrict__ gamma, const float* __restrict__ beta,
                         float* __restrict__ xf, u16t* __restrict__ xb, long n) {
  const float sc = rsqrtf(1.0f + 1e-5f);
  long i = (long)blockIdx.x * blockDim.x + threadIdx.x;
  long st = (long)gridDim.x * blockDim.x;
  for (; i < n; i += st) {
    int c = (int)(i & 1023);
    float v = acc[i] + gb[c] + gb[1024 + c] + gb[2048 + c];
    v = v * (gamma[c] * sc) + beta[c];
    v = v > 0.f ? v : expm1f(v);
    if (xf) xf[i] = v;
    if (xb) xb[i] = f2bf(v);
  }
}

// ---------------------------------------------------------------------------
// Host-side orchestration
// ---------------------------------------------------------------------------
namespace {

struct WS {
  u16t *xbf, *swb, *e1wb, *e2wb, *gwb, *xtb;
  float *outb, *x1, *x2, *mT, *sT, *tmpT, *hsrc, *hdst, *asrc, *adst, *m2, *s2, *ebuf, *gap;
};

inline int gridFor(long n) { long b = (n + 255) / 256; if (b > (1L << 20)) b = (1L << 20); return (int)b; }

inline void gemm(hipStream_t st, const u16t* A, const u16t* Bt, const float* bias,
                 float* C, int M, int N, int K) {
  dim3 grid(N / BN, (M + BM - 1) / BM);
  k_gemm_bf16<<<grid, 256, 0, st>>>(A, Bt, bias, C, M, N, K);
}
inline void f2bfA(hipStream_t st, const float* s, u16t* d, long n) {
  k_f2bf<<<gridFor(n), 256, 0, st>>>(s, d, n);
}
inline void f2bfT(hipStream_t st, const float* s, u16t* d, int K, int N) {
  k_f2bf_t<<<gridFor((long)K * N), 256, 0, st>>>(s, d, K, N);
}
inline void fillA(hipStream_t st, void* p, unsigned v, long n) {
  k_fill_u32<<<gridFor(n), 256, 0, st>>>((unsigned*)p, v, n);
}

void run_layer(hipStream_t st, const WS& w, const u16t* xbf,
               int nsrc, int nt, int cin, int E,
               const int* const* row, const int* const* col,
               void* const* P, const int* years,
               float* xf_out, u16t* xbf_out)
{
  const float* sW    = (const float*)P[0];
  const float* sb    = (const float*)P[1];
  const float* e1W   = (const float*)P[2];
  const float* e1b   = (const float*)P[3];
  const float* e2W   = (const float*)P[4];
  const float* e2b   = (const float*)P[5];
  const float* att1  = (const float*)P[6];
  const float* att2  = (const float*)P[7];
  const float* gW    = (const float*)P[8];
  const float* gas   = (const float*)P[9];
  const float* gad   = (const float*)P[10];
  const float* gb    = (const float*)P[11];
  const float* gamma = (const float*)P[12];
  const float* beta  = (const float*)P[13];

  // weights -> bf16, transposed to [N,K] for vectorized LDS staging
  f2bfT(st, sW,  w.swb,  cin, 1024);
  f2bfT(st, e1W, w.e1wb, cin, cin);
  f2bfT(st, e2W, w.e2wb, cin, cin);
  for (int j = 0; j < 3; ++j)
    f2bfT(st, gW + (size_t)j * cin * 1024, w.gwb + (size_t)j * cin * 1024, cin, 1024);

  // skip + temporal projections
  gemm(st, xbf, w.swb,  sb,  w.outb, nt,   1024, cin);
  gemm(st, xbf, w.e1wb, e1b, w.x1,   nsrc, cin,  cin);
  gemm(st, xbf, w.e2wb, e2b, w.x2,   nt,   cin,  cin);

  // temporal attention (relation-0 edges)
  k_gap<<<(E + 255) / 256, 256, 0, st>>>(row[0], col[0], years, w.gap, E);
  fillA(st, w.mT,   0xFF800000u, (long)nt * cin);
  fillA(st, w.sT,   0u,          (long)nt * cin);
  fillA(st, w.tmpT, 0u,          (long)nt * cin);
  k_temporal_max<<<E, 256, 0, st>>>(w.x1, w.x2, att1, att2, row[0], col[0], w.gap, w.mT, cin);
  k_temporal_sum<<<E, 256, 0, st>>>(w.x1, w.x2, att1, att2, row[0], col[0], w.gap, w.mT, w.sT, cin);
  k_temporal_agg<<<E, 256, 0, st>>>(w.x1, w.x2, att1, att2, row[0], col[0], w.gap, w.mT, w.sT, w.tmpT, cin);
  f2bfA(st, w.tmpT, w.xtb, (long)nt * cin);

  // three relation GATs accumulate into outb
  for (int j = 0; j < 3; ++j) {
    const u16t* gwj = w.gwb + (size_t)j * cin * 1024;
    gemm(st, xbf,   gwj, nullptr, w.hsrc, nsrc, 1024, cin);
    gemm(st, w.xtb, gwj, nullptr, w.hdst, nt,   1024, cin);
    k_attn_reduce<<<nsrc, 256, 0, st>>>(w.hsrc, gas + j * 1024, w.asrc);
    k_attn_reduce<<<nt,   256, 0, st>>>(w.hdst, gad + j * 1024, w.adst);
    fillA(st, w.m2, 0xFF800000u, (long)nt * 4);
    fillA(st, w.s2, 0u,          (long)nt * 4);
    k_gat_logits<<<(E * 4 + 255) / 256, 256, 0, st>>>(w.asrc, w.adst, row[j], col[j], w.ebuf, w.m2, E);
    k_gat_exp<<<(E * 4 + 255) / 256, 256, 0, st>>>(w.ebuf, row[j], w.m2, w.s2, E);
    k_gat_scatter<<<E, 256, 0, st>>>(w.hsrc, w.ebuf, w.s2, row[j], col[j], w.outb, E);
  }

  // bias sum + BN scale + ELU, emit next-layer bf16 and/or final f32
  k_bn_elu<<<gridFor((long)nt * 1024), 256, 0, st>>>(w.outb, gb, gamma, beta, xf_out, xbf_out, (long)nt * 1024);
}

} // namespace

extern "C" void kernel_launch(void* const* d_in, const int* in_sizes, int n_in,
                              void* d_out, int out_size, void* d_ws, size_t ws_size,
                              hipStream_t stream)
{
  (void)in_sizes; (void)n_in; (void)out_size; (void)ws_size;

  constexpr int N0 = 60000, N1 = 15000, N2 = 4000;
  constexpr int E0 = 120000, E1 = 60000;
  constexpr int CIN = 768;

  const float* x0  = (const float*)d_in[0];
  const int* years = (const int*)d_in[1];
  const int* row[2][3]; const int* col[2][3];
  for (int j = 0; j < 3; ++j) {          // setup_inputs dict order: row0_j, col0_j, row1_j, col1_j
    row[0][j] = (const int*)d_in[2 + 4 * j];
    col[0][j] = (const int*)d_in[3 + 4 * j];
    row[1][j] = (const int*)d_in[4 + 4 * j];
    col[1][j] = (const int*)d_in[5 + 4 * j];
  }

  // carve workspace (sized for the larger of both layers)
  char* p = (char*)d_ws;
  auto alloc = [&](size_t bytes) -> void* {
    void* r = (void*)p; p += (bytes + 255) & ~(size_t)255; return r;
  };
  WS w;
  w.xbf  = (u16t*) alloc((size_t)N0 * CIN * 2);        // 60000x768 bf16 (reused as layer-1 x)
  w.swb  = (u16t*) alloc((size_t)1024 * 1024 * 2);
  w.e1wb = (u16t*) alloc((size_t)1024 * 1024 * 2);
  w.e2wb = (u16t*) alloc((size_t)1024 * 1024 * 2);
  w.gwb  = (u16t*) alloc((size_t)3 * 1024 * 1024 * 2);
  w.xtb  = (u16t*) alloc((size_t)N1 * CIN * 2);
  w.outb = (float*)alloc((size_t)N1 * 1024 * 4);
  w.x1   = (float*)alloc((size_t)N0 * CIN * 4);
  w.x2   = (float*)alloc((size_t)N1 * CIN * 4);
  w.mT   = (float*)alloc((size_t)N1 * CIN * 4);
  w.sT   = (float*)alloc((size_t)N1 * CIN * 4);
  w.tmpT = (float*)alloc((size_t)N1 * CIN * 4);
  w.hsrc = (float*)alloc((size_t)N0 * 1024 * 4);
  w.hdst = (float*)alloc((size_t)N1 * 1024 * 4);
  w.asrc = (float*)alloc((size_t)N0 * 4 * 4);
  w.adst = (float*)alloc((size_t)N1 * 4 * 4);
  w.m2   = (float*)alloc((size_t)N1 * 4 * 4);
  w.s2   = (float*)alloc((size_t)N1 * 4 * 4);
  w.ebuf = (float*)alloc((size_t)E0 * 4 * 4);
  w.gap  = (float*)alloc((size_t)E0 * 4);

  // layer 0: convert input x to bf16, run, emit bf16 activations back into xbf
  f2bfA(stream, x0, w.xbf, (long)N0 * CIN);
  run_layer(stream, w, w.xbf, N0, N1, CIN, E0, row[0], col[0],
            d_in + 14, years, nullptr, w.xbf);

  // layer 1: reads bf16 activations, writes final f32 result straight to d_out
  run_layer(stream, w, w.xbf, N1, N2, 1024, E1, row[1], col[1],
            d_in + 28, years, (float*)d_out, nullptr);
}
